// LocalAttentionDeformable_51015621542050
// MI455X (gfx1250) — matrix operand order repaired and checked by reading, not compiled, not run on hardware
//
#include <hip/hip_runtime.h>
#include <hip/hip_bf16.h>

typedef __attribute__((ext_vector_type(16))) _Float16 v16h;
typedef __attribute__((ext_vector_type(8)))  _Float16 v8h;
typedef __attribute__((ext_vector_type(4)))  _Float16 v4h;
typedef __attribute__((ext_vector_type(2)))  _Float16 v2h;
typedef __attribute__((ext_vector_type(8)))  float    v8f;

// ---------------------------------------------------------------------------
// Problem constants (from reference)
// ---------------------------------------------------------------------------
#define BATCH      32
#define C_IN       384
#define HGT        56
#define WID        56
#define HW_PLANE   (HGT * WID)              // 3136
#define WH         7
#define WW         7
#define N_TOK      49
#define HEADS      8
#define HD         64
#define INNER      512
#define QKV_COLS   1536
#define N_WIN      2048
#define M_TOTAL    (N_WIN * N_TOK)          // 100352, divisible by 64
#define SCALE_Q    0.125f
#define OUT_ELEMS  (BATCH * C_IN * HGT * WID)

// ---------------------------------------------------------------------------
// gfx1250 async global->LDS copy (ASYNCcnt path, ISA 10 / 15.18.3 op 98)
// ---------------------------------------------------------------------------
__device__ __forceinline__ unsigned lds_off32(const void* p) {
    // LDS aperture: generic addr low 32 bits == LDS byte offset
    return (unsigned)(unsigned long long)p;
}
__device__ __forceinline__ void async_load_b128(unsigned lds_byte, const void* gaddr) {
    asm volatile("global_load_async_to_lds_b128 %0, %1, off"
                 :: "v"(lds_byte), "v"(gaddr) : "memory");
}
__device__ __forceinline__ void wait_async0() {
    asm volatile("s_wait_asynccnt 0" ::: "memory");
}

// ---------------------------------------------------------------------------
// WMMA fragment loaders — 2x ds_load_b128 per fragment (wave32, ISA 7.12.2)
// ---------------------------------------------------------------------------
__device__ __forceinline__ v16h load_frag_a(const _Float16* src, int stride) {
    const int lane = threadIdx.x & 31;
    const _Float16* p = src + (lane & 15) * stride + ((lane >> 4) << 3);
    union { v16h v; v8h h[2]; } u;
    u.h[0] = *(const v8h*)(p);        // K = kb .. kb+7
    u.h[1] = *(const v8h*)(p + 16);   // K = kb+16 .. kb+23
    return u.v;
}
// B 32x16 from K-contiguous (transposed) storage: rows are N, cols K.
__device__ __forceinline__ v16h load_frag_b(const _Float16* src, int stride) {
    const int lane = threadIdx.x & 31;
    const _Float16* p = src + (lane & 15) * stride + ((lane >> 4) << 3);
    union { v16h v; v8h h[2]; } u;
    u.h[0] = *(const v8h*)(p);
    u.h[1] = *(const v8h*)(p + 16);
    return u.v;
}
// C/D layout: VGPR r -> m = r + ((lane>>4)<<3), n = lane & 15
#define C_M_OFF(r)  ((r) + (((threadIdx.x & 31) >> 4) << 3))
#define C_N_OFF()   ((threadIdx.x & 31) & 15)

// stage a 32(K)x64(N) f32 weight tile into K-contiguous f16 LDS (one 4x4
// block per thread: 4x global_load_b128 -> register transpose -> 4x b64 store)
__device__ __forceinline__ void stage_w_tile(_Float16 (*Bst)[40],
                                             const float* __restrict__ w,
                                             long row_stride, int k0, int n0) {
    const int tid = threadIdx.x;
    const int nn0 = (tid & 15) * 4;
    const int kkb = (tid >> 4) * 4;
    float blk[4][4];
#pragma unroll
    for (int kj = 0; kj < 4; ++kj) {
        const float4 t = *(const float4*)&w[(long)(k0 + kkb + kj) * row_stride + (n0 + nn0)];
        blk[kj][0] = t.x; blk[kj][1] = t.y; blk[kj][2] = t.z; blk[kj][3] = t.w;
    }
#pragma unroll
    for (int nj = 0; nj < 4; ++nj) {
        v4h h;
        h[0] = (_Float16)blk[0][nj];
        h[1] = (_Float16)blk[1][nj];
        h[2] = (_Float16)blk[2][nj];
        h[3] = (_Float16)blk[3][nj];
        *(v4h*)&Bst[nn0 + nj][kkb] = h;
    }
}

// ===========================================================================
// Kernel 1: fused window-gather + QKV projection. Block tile 64M x 64N,
// 4 waves, 4 accumulators/wave. grid: (M_TOTAL/64, 1536/64)
// ===========================================================================
__global__ void __launch_bounds__(128)
swin_qkv_kernel(const float* __restrict__ x,
                const float* __restrict__ w_qkv,
                const float* __restrict__ b_qkv,
                _Float16* __restrict__ qws,
                _Float16* __restrict__ kws,
                _Float16* __restrict__ vtws) {
    __shared__ __attribute__((aligned(16))) _Float16 As[64][32];
    __shared__ __attribute__((aligned(16))) _Float16 Bst[64][40];  // transposed [n][k]

    const int tid  = threadIdx.x;
    const int wave = tid >> 5;
    const int lane = tid & 31;
    const int m0   = blockIdx.x * 64;
    const int n0   = blockIdx.y * 64;

    // Hoisted gather base: thread owns A row ar, channel half [akb, akb+16)
    const int ar  = tid & 63;
    const int akb = (tid >> 6) << 4;    // 0 or 16
    long abase;
    {
        const int m   = m0 + ar;
        const int bm  = m / N_TOK;
        const int tok = m - bm * N_TOK;
        const int b   = bm >> 6, rr = bm & 63;
        const int r1  = rr >> 3, r2 = rr & 7;
        const int h1  = tok / WW, w1 = tok - h1 * WW;
        abase = (long)b * C_IN * HW_PLANE + (long)(r1 * WH + h1) * WID + (r2 * WW + w1);
    }

    v8f acc[4] = {};

    for (int k0 = 0; k0 < C_IN; k0 += 32) {
        if (k0 + 32 < C_IN)
            __builtin_prefetch(&w_qkv[(long)(k0 + 32) * QKV_COLS + n0 + lane], 0, 1);

        // stage A: thread's fixed row, 16 channels, packed pair stores
#pragma unroll
        for (int kk = 0; kk < 16; kk += 2) {
            v2h p;
            p[0] = (_Float16)x[abase + (long)(k0 + akb + kk)     * HW_PLANE];
            p[1] = (_Float16)x[abase + (long)(k0 + akb + kk + 1) * HW_PLANE];
            *(v2h*)&As[ar][akb + kk] = p;
        }
        // stage B 32x64 transposed (vectorized)
        stage_w_tile(Bst, w_qkv, QKV_COLS, k0, n0);
        __syncthreads();

        const v16h a = load_frag_a(&As[wave * 16][0], 32);
#pragma unroll
        for (int nt = 0; nt < 4; ++nt) {
            const v16h bb = load_frag_b(&Bst[nt * 16][0], 40);
            acc[nt] = __builtin_amdgcn_wmma_f32_16x16x32_f16(false, a, false, bb,
                                                             (short)0, acc[nt], false, false);
        }
        __syncthreads();
    }

    // scatter into q/k/vt workspace (f16), bias + q-scale applied
#pragma unroll
    for (int nt = 0; nt < 4; ++nt) {
#pragma unroll
        for (int r = 0; r < 8; ++r) {
            const int m = m0 + wave * 16 + C_M_OFF(r);
            const int j = n0 + nt * 16 + C_N_OFF();
            const float val = acc[nt][r] + b_qkv[j];
            const int bm    = m / N_TOK;
            const int tok   = m - bm * N_TOK;
            const int which = j >> 9;                // 0=q 1=k 2=v
            const int head  = (j >> 6) & 7;
            const int d     = j & 63;
            const long bmh  = (long)bm * HEADS + head;
            if (which == 0)
                qws[(bmh * N_TOK + tok) * HD + d] = (_Float16)(val * SCALE_Q);
            else if (which == 1)
                kws[(bmh * N_TOK + tok) * HD + d] = (_Float16)val;
            else  // v stored transposed: [bmh][d][64 tokens]
                vtws[(bmh * HD + d) * 64 + tok] = (_Float16)val;
        }
    }
}

// ===========================================================================
// Kernel 2: fused attention per (window, head). Async-to-LDS staging.
// grid (N_WIN, HEADS), 128 thr.
// ===========================================================================
__global__ void __launch_bounds__(128)
swin_attn_kernel(const _Float16* __restrict__ qws,
                 const _Float16* __restrict__ kws,
                 const _Float16* __restrict__ vtws,
                 const float* __restrict__ bias_table,
                 float* __restrict__ attn_out,
                 _Float16* __restrict__ ohws) {
    __shared__ __attribute__((aligned(16))) _Float16 qs[64][64];   // q, later P
    __shared__ __attribute__((aligned(16))) _Float16 ks[64][64];
    __shared__ __attribute__((aligned(16))) _Float16 vts[64][64];  // [d][token]
    __shared__ float Ss[64][65];

    const int bm   = blockIdx.x;
    const int head = blockIdx.y;
    const int tid  = threadIdx.x;
    const int wave = tid >> 5;
    const long bmh = (long)bm * HEADS + head;

    const char* qg  = (const char*)(qws  + bmh * N_TOK * HD);
    const char* kg  = (const char*)(kws  + bmh * N_TOK * HD);
    const char* vtg = (const char*)(vtws + bmh * HD * 64);

    const unsigned qb = lds_off32(&qs[0][0]);
    const unsigned kb = lds_off32(&ks[0][0]);
    const unsigned vb = lds_off32(&vts[0][0]);

    // async copy: q,k = 49*64 halves = 392 x b128 ; vt = 512 x b128
    for (int i = tid; i < 392; i += 128) {
        async_load_b128(qb + i * 16, qg + i * 16);
        async_load_b128(kb + i * 16, kg + i * 16);
    }
    for (int i = tid; i < 512; i += 128) {
        async_load_b128(vb + i * 16, vtg + i * 16);
    }
    wait_async0();
    // zero the padding: q/k rows 49..63, vt columns 49..63
    for (int i = tid; i < 15 * 64; i += 128) {
        const int r = 49 + (i >> 6), c = i & 63;
        qs[r][c] = (_Float16)0.0f;
        ks[r][c] = (_Float16)0.0f;
    }
    for (int i = tid; i < 64 * 15; i += 128) {
        const int r = i / 15, c = 49 + (i % 15);
        vts[r][c] = (_Float16)0.0f;
    }
    __syncthreads();

    // ---- S = q @ k^T (+ relative position bias) ----
#pragma unroll
    for (int t = 0; t < 4; ++t) {
        const int tile = wave * 4 + t;
        const int mt = tile >> 2, nt = tile & 3;
        v8f c = {};
#pragma unroll
        for (int kk0 = 0; kk0 < HD; kk0 += 32) {
            const v16h a  = load_frag_a(&qs[mt * 16][kk0], 64);
            const v16h bb = load_frag_b(&ks[nt * 16][kk0], 64);
            c = __builtin_amdgcn_wmma_f32_16x16x32_f16(false, a, false, bb,
                                                       (short)0, c, false, false);
        }
#pragma unroll
        for (int r = 0; r < 8; ++r) {
            const int m = mt * 16 + C_M_OFF(r);
            const int n = nt * 16 + C_N_OFF();
            float sv;
            if (m < N_TOK && n < N_TOK) {
                const int mh = m / WW, mw = m - mh * WW;
                const int nh = n / WW, nw = n - nh * WW;
                const int idx = (mh - nh + WH - 1) * (2 * WW - 1) + (mw - nw + WW - 1);
                sv = c[r] + bias_table[idx * HEADS + head];
            } else {
                sv = -1e30f;
            }
            Ss[m][n] = sv;
        }
    }
    __syncthreads();

    // ---- f32 row softmax; emit attn output; re-quantize P into qs ----
    for (int m = tid; m < N_TOK; m += 128) {
        float mx = -1e30f;
        for (int n = 0; n < N_TOK; ++n) mx = fmaxf(mx, Ss[m][n]);
        float sum = 0.0f;
        for (int n = 0; n < N_TOK; ++n) {
            const float e = __expf(Ss[m][n] - mx);
            Ss[m][n] = e;
            sum += e;
        }
        const float inv = 1.0f / sum;
        float* ao = attn_out + (bmh * N_TOK + m) * N_TOK;
        for (int n = 0; n < N_TOK; ++n) {
            const float p = Ss[m][n] * inv;
            ao[n]    = p;
            qs[m][n] = (_Float16)p;
        }
        for (int n = N_TOK; n < 64; ++n) qs[m][n] = (_Float16)0.0f;
    }
    __syncthreads();

    // ---- O = P @ V (vts is K-contiguous per dim row) ----
#pragma unroll
    for (int t = 0; t < 4; ++t) {
        const int tile = wave * 4 + t;
        const int mt = tile >> 2, nt = tile & 3;
        v8f c = {};
#pragma unroll
        for (int kk0 = 0; kk0 < 64; kk0 += 32) {
            const v16h a  = load_frag_a(&qs[mt * 16][kk0], 64);
            const v16h bb = load_frag_b(&vts[nt * 16][kk0], 64);
            c = __builtin_amdgcn_wmma_f32_16x16x32_f16(false, a, false, bb,
                                                       (short)0, c, false, false);
        }
#pragma unroll
        for (int r = 0; r < 8; ++r) {
            const int m = mt * 16 + C_M_OFF(r);
            const int d = nt * 16 + C_N_OFF();
            if (m < N_TOK) {
                ohws[((long)bm * N_TOK + m) * INNER + head * HD + d] = (_Float16)c[r];
            }
        }
    }
}

// ===========================================================================
// Kernel 3: output projection + reverse window partition.
// A (f16) staged via async-to-LDS; B vector-staged; 64x64 block tile.
// grid: (M_TOTAL/64, 384/64)
// ===========================================================================
__global__ void __launch_bounds__(128)
swin_proj_kernel(const _Float16* __restrict__ ohws,
                 const float* __restrict__ w_out,
                 const float* __restrict__ b_out,
                 float* __restrict__ out) {
    __shared__ __attribute__((aligned(16))) _Float16 As[64][32];
    __shared__ __attribute__((aligned(16))) _Float16 Bst[64][40];

    const int tid  = threadIdx.x;
    const int wave = tid >> 5;
    const int lane = tid & 31;
    const int m0   = blockIdx.x * 64;
    const int n0   = blockIdx.y * 64;

    const unsigned abase = lds_off32(&As[0][0]);
    v8f acc[4] = {};

    for (int k0 = 0; k0 < INNER; k0 += 32) {
        if (k0 + 32 < INNER)
            __builtin_prefetch(&w_out[(long)(k0 + 32) * C_IN + n0 + lane], 0, 1);

        // async-stage A: 64 rows x 32 halves = 64B/row = 4 chunks/row
        for (int i = tid; i < 64 * 4; i += 128) {
            const int r = i >> 2, ch = i & 3;
            const char* g = (const char*)(ohws + (long)(m0 + r) * INNER + k0) + ch * 16;
            async_load_b128(abase + r * 64 + ch * 16, g);
        }
        // stage B 32x64 transposed (vectorized)
        stage_w_tile(Bst, w_out, C_IN, k0, n0);
        wait_async0();
        __syncthreads();

        const v16h a = load_frag_a(&As[wave * 16][0], 32);
#pragma unroll
        for (int nt = 0; nt < 4; ++nt) {
            const v16h bb = load_frag_b(&Bst[nt * 16][0], 40);
            acc[nt] = __builtin_amdgcn_wmma_f32_16x16x32_f16(false, a, false, bb,
                                                             (short)0, acc[nt], false, false);
        }
        __syncthreads();
    }

    // bias + reverse window partition scatter into NCHW
#pragma unroll
    for (int nt = 0; nt < 4; ++nt) {
#pragma unroll
        for (int r = 0; r < 8; ++r) {
            const int m = m0 + wave * 16 + C_M_OFF(r);
            const int j = n0 + nt * 16 + C_N_OFF();
            const float val = acc[nt][r] + b_out[j];
            const int bm  = m / N_TOK;
            const int tok = m - bm * N_TOK;
            const int b   = bm >> 6;
            const int rr  = bm & 63;
            const int r1  = rr >> 3, r2 = rr & 7;
            const int h1  = tok / WW, w1 = tok - h1 * WW;
            out[(((long)b * C_IN + j) * HGT + (r1 * WH + h1)) * WID + (r2 * WW + w1)] = val;
        }
    }
}

// ===========================================================================
extern "C" void kernel_launch(void* const* d_in, const int* in_sizes, int n_in,
                              void* d_out, int out_size, void* d_ws, size_t ws_size,
                              hipStream_t stream) {
    const float* x          = (const float*)d_in[0];
    const float* w_qkv      = (const float*)d_in[1];
    const float* b_qkv      = (const float*)d_in[2];
    const float* w_out      = (const float*)d_in[3];
    const float* b_out      = (const float*)d_in[4];
    const float* bias_table = (const float*)d_in[5];

    float* out      = (float*)d_out;
    float* attn_out = out + (long)OUT_ELEMS;

    // workspace: q,k [bmh][49][64] f16 ; vt [bmh][64][64] f16 ; oh [M][512] f16
    const size_t QK_BYTES = (size_t)N_WIN * HEADS * N_TOK * HD * sizeof(_Float16);
    const size_t VT_BYTES = (size_t)N_WIN * HEADS * HD * 64 * sizeof(_Float16);
    char* ws = (char*)d_ws;
    _Float16* qws  = (_Float16*)(ws);
    _Float16* kws  = (_Float16*)(ws + QK_BYTES);
    _Float16* vtws = (_Float16*)(ws + 2 * QK_BYTES);
    _Float16* ohws = (_Float16*)(ws + 2 * QK_BYTES + VT_BYTES);

    (void)in_sizes; (void)n_in; (void)out_size; (void)ws_size;

    swin_qkv_kernel <<<dim3(M_TOTAL / 64, QKV_COLS / 64), 128, 0, stream>>>(
        x, w_qkv, b_qkv, qws, kws, vtws);

    swin_attn_kernel<<<dim3(N_WIN, HEADS), 128, 0, stream>>>(
        qws, kws, vtws, bias_table, attn_out, ohws);

    swin_proj_kernel<<<dim3(M_TOTAL / 64, C_IN / 64), 128, 0, stream>>>(
        ohws, w_out, b_out, out);
}
